// DomainSpecificAttention_17394617548910
// MI455X (gfx1250) — matrix operand order repaired and checked by reading, not compiled, not run on hardware
//
#include <hip/hip_runtime.h>

typedef _Float16 f16;
typedef _Float16 v8h  __attribute__((ext_vector_type(8)));
typedef _Float16 v16h __attribute__((ext_vector_type(16)));
typedef float    v8f  __attribute__((ext_vector_type(8)));

#define BB 2
#define HH 48
#define WW 48
#define CC 64
#define NTOK 2304      // H*W
#define NHD 8
#define KDIM 64
#define NBANDS 8
#define PI_F 3.14159265358979323846f

// ---------------- workspace layout (bytes) ----------------
constexpr size_t o_dct = 0;                        // 48*48 f32
constexpr size_t o_q   = o_dct + 9216;             // B*N*64 f32
constexpr size_t o_qh  = o_q   + 1179648;          // B*NH*N*64 f16
constexpr size_t o_kh  = o_qh  + 4718592;
constexpr size_t o_vh  = o_kh  + 4718592;          // transposed: [b][h][d][n]
constexpr size_t o_o   = o_vh  + 4718592;          // B*N*NH*64 f32
constexpr size_t o_t1  = o_o   + 9437184;          // B*H*W*C f32 temp
constexpr size_t o_t2  = o_t1  + 1179648;          // dct result / wd
constexpr size_t o_t3  = o_t2  + 1179648;          // freq conv out
constexpr size_t o_bf  = o_t3  + 1179648;          // 2*8*64 f32
constexpr size_t o_bw  = o_bf  + 4096;             // 2*8 f32
constexpr size_t o_pp  = o_bw  + 256;              // 144 f32 partials
constexpr size_t o_pr  = o_pp  + 768;              // 2 f32
constexpr size_t o_ll0 = o_pr  + 256;              // 2*48*48
constexpr size_t o_ll1 = o_ll0 + 18432;            // 2*24*24
constexpr size_t o_h1  = o_ll1 + 4608;
constexpr size_t o_v1  = o_h1  + 4608;
constexpr size_t o_d1  = o_v1  + 4608;
constexpr size_t o_ll2 = o_d1  + 4608;             // 2*12*12
constexpr size_t o_h2  = o_ll2 + 1152;
constexpr size_t o_v2  = o_h2  + 1152;
constexpr size_t o_d2  = o_v2  + 1152;
constexpr size_t o_ll3 = o_d2  + 1152;             // 2*6*6
constexpr size_t o_h3  = o_ll3 + 288;
constexpr size_t o_v3  = o_h3  + 288;
constexpr size_t o_d3  = o_v3  + 288;
constexpr size_t o_r6  = o_d3  + 288;
constexpr size_t o_r12 = o_r6  + 288;
constexpr size_t o_r24 = o_r12 + 1152;
constexpr size_t o_lg1 = o_r24 + 4608;             // 2*3 logits each
constexpr size_t o_lg2 = o_lg1 + 256;
constexpr size_t o_lg3 = o_lg2 + 256;
constexpr size_t o_yp  = o_lg3 + 256;              // B*H*W*64 f32

// ---------------- small helpers ----------------
__device__ inline v16h pack16(const f16* p0, const f16* p1) {
  v8h a = *(const v8h*)p0;
  v8h b = *(const v8h*)p1;
  v16h r;
#pragma unroll
  for (int i = 0; i < 8; i++) { r[i] = a[i]; r[i + 8] = b[i]; }
  return r;
}

__device__ inline float reduce64(float v, float* red, int t) {
  red[t] = v; __syncthreads();
#pragma unroll
  for (int off = 32; off >= 1; off >>= 1) {
    if (t < off) red[t] += red[t + off];
    __syncthreads();
  }
  float r = red[0]; __syncthreads();
  return r;
}

// ---------------- DCT matrix ----------------
__global__ void k_dctmat(float* D) {
  int tid = blockIdx.x * blockDim.x + threadIdx.x;
  if (tid >= 48 * 48) return;
  int k = tid / 48, n = tid % 48;
  double s = (k == 0) ? 0.14433756729740643 : 0.20412414523193150; // sqrt(1/48), sqrt(2/48)
  D[tid] = (float)(s * cos(M_PI * (2.0 * n + 1.0) * k / 96.0));
}

// ---------------- pos 1x1 conv: q = x @ pos_W + pos_b ----------------
__global__ void k_pos(const float* __restrict__ x, const float* __restrict__ W,
                      const float* __restrict__ bias, float* __restrict__ q) {
  int tid = blockIdx.x * blockDim.x + threadIdx.x;
  if (tid >= BB * NTOK * KDIM) return;
  int d = tid & 63;
  size_t row = (size_t)(tid >> 6);
  const float* xr = x + row * CC;
  float acc = bias[d];
#pragma unroll 4
  for (int e = 0; e < CC; e++) acc += xr[e] * W[e * KDIM + d];
  q[tid] = acc;
}

// ---------------- head projection: out[b][h][n][d] (f16, optional scale) ----------------
__global__ void k_proj(const float* __restrict__ in, const float* __restrict__ W,
                       const float* __restrict__ bias, f16* __restrict__ out, float scale) {
  int tid = blockIdx.x * blockDim.x + threadIdx.x;
  if (tid >= BB * NHD * NTOK * KDIM) return;
  int d = tid & 63;
  int n = (tid >> 6) % NTOK;
  int h = ((tid >> 6) / NTOK) % NHD;
  int b = (tid >> 6) / (NTOK * NHD);
  const float* row = in + ((size_t)b * NTOK + n) * 64;
  float acc = bias[h * 64 + d];
#pragma unroll 4
  for (int e = 0; e < 64; e++) acc += row[e] * W[(e * NHD + h) * 64 + d];
  out[tid] = (f16)(acc * scale);
}

// ---------------- head projection, transposed out: out[b][h][d][n] (f16) ----------------
__global__ void k_projT(const float* __restrict__ in, const float* __restrict__ W,
                        const float* __restrict__ bias, f16* __restrict__ out) {
  int tid = blockIdx.x * blockDim.x + threadIdx.x;
  if (tid >= BB * NHD * KDIM * NTOK) return;
  int n = tid % NTOK;
  int d = (tid / NTOK) & 63;
  int h = (tid / (NTOK * KDIM)) % NHD;
  int b = tid / (NTOK * KDIM * NHD);
  const float* row = in + ((size_t)b * NTOK + n) * 64;
  float acc = bias[h * 64 + d];
#pragma unroll 4
  for (int e = 0; e < 64; e++) acc += row[e] * W[(e * NHD + h) * 64 + d];
  out[tid] = (f16)acc;
}

// ---------------- flash attention (WMMA f16 -> f32), 64-key blocks ----------------
__global__ __launch_bounds__(128) void k_flash_attn(const f16* __restrict__ qh,
                                                    const f16* __restrict__ kh,
                                                    const f16* __restrict__ vhT,
                                                    float* __restrict__ o) {
  __shared__ __align__(32) f16 Ksh[64 * 72];     // [key][hd], pad 72
  __shared__ __align__(32) f16 VshT[64 * 72];    // [hd][key], pad 72
  __shared__ __align__(32) f16 Psh[4 * 16 * 72]; // per-wave P staging [row][key]

  int bh   = blockIdx.x / 36;   // b*NH + h
  int qblk = blockIdx.x % 36;
  int b = bh / NHD, h = bh % NHD;
  int tid = threadIdx.x;
  int wave = tid >> 5, lane = tid & 31;
  int g = lane >> 4, ln = lane & 15;

  const f16* qb = qh  + (size_t)bh * NTOK * KDIM;
  const f16* kb = kh  + (size_t)bh * NTOK * KDIM;
  const f16* vb = vhT + (size_t)bh * KDIM * NTOK;   // [d][n]
  int q0 = qblk * 64 + wave * 16;

  // Q A-fragments (two hd-chunks of 32); 1/sqrt(KD) already folded into qh
  const f16* qrow = qb + (size_t)(q0 + ln) * KDIM;
  v16h Aq0 = pack16(qrow + g * 16,      qrow + g * 16 + 8);
  v16h Aq1 = pack16(qrow + 32 + g * 16, qrow + 32 + g * 16 + 8);

  v8f O[4] = {v8f{}, v8f{}, v8f{}, v8f{}};
  float mrow[8], lrow[8];
#pragma unroll
  for (int v = 0; v < 8; v++) { mrow[v] = -1e30f; lrow[v] = 0.f; }
  f16* Pw = Psh + wave * 16 * 72;

  // staging indices: 128 threads cover 64 rows x 64 halves (32 halves/thread)
  int sr = tid >> 1, sseg = tid & 1;

  for (int j = 0; j < NTOK / 64; j++) {
    int kb0 = j * 64;
    // cooperative staging: K row-major [key][hd], V already transposed [hd][key]
    {
      const f16* ks = kb + (size_t)(kb0 + sr) * KDIM + sseg * 32;
      f16* kd = Ksh + sr * 72 + sseg * 32;
      *(v8h*)(kd)      = *(const v8h*)(ks);
      *(v8h*)(kd + 8)  = *(const v8h*)(ks + 8);
      *(v8h*)(kd + 16) = *(const v8h*)(ks + 16);
      *(v8h*)(kd + 24) = *(const v8h*)(ks + 24);
      const f16* vs = vb + (size_t)sr * NTOK + kb0 + sseg * 32;
      f16* vd = VshT + sr * 72 + sseg * 32;
      *(v8h*)(vd)      = *(const v8h*)(vs);
      *(v8h*)(vd + 8)  = *(const v8h*)(vs + 8);
      *(v8h*)(vd + 16) = *(const v8h*)(vs + 16);
      *(v8h*)(vd + 24) = *(const v8h*)(vs + 24);
    }
    if (j < NTOK / 64 - 1) {  // prefetch next tiles (uniform branch)
      __builtin_prefetch(kb + (size_t)(kb0 + 64 + sr) * KDIM, 0, 1);
      __builtin_prefetch(vb + (size_t)sr * NTOK + kb0 + 64, 0, 1);
    }
    __syncthreads();

    // S = Q K^T : four 16-key sub-tiles
    v8f S[4];
#pragma unroll
    for (int s = 0; s < 4; s++) {
      v8f acc = {};
      v16h Bk = pack16(Ksh + (s * 16 + ln) * 72 + g * 16,
                       Ksh + (s * 16 + ln) * 72 + g * 16 + 8);
      acc = __builtin_amdgcn_wmma_f32_16x16x32_f16(false, Aq0, false, Bk, (short)0, acc, false, false);
      Bk = pack16(Ksh + (s * 16 + ln) * 72 + 32 + g * 16,
                  Ksh + (s * 16 + ln) * 72 + 32 + g * 16 + 8);
      acc = __builtin_amdgcn_wmma_f32_16x16x32_f16(false, Aq1, false, Bk, (short)0, acc, false, false);
      S[s] = acc;
    }

    // online softmax per row (rows live in 16-lane halves), one update per 64 keys
#pragma unroll
    for (int v = 0; v < 8; v++) {
      float a0 = S[0][v], a1 = S[1][v], a2 = S[2][v], a3 = S[3][v];
      float mx = fmaxf(fmaxf(a0, a1), fmaxf(a2, a3));
#pragma unroll
      for (int off = 8; off >= 1; off >>= 1) mx = fmaxf(mx, __shfl_xor(mx, off, 32));
      float mn = fmaxf(mrow[v], mx);
      float alpha = __expf(mrow[v] - mn);
      mrow[v] = mn;
      float p0 = __expf(a0 - mn), p1 = __expf(a1 - mn);
      float p2 = __expf(a2 - mn), p3 = __expf(a3 - mn);
      float rs = (p0 + p1) + (p2 + p3);
#pragma unroll
      for (int off = 8; off >= 1; off >>= 1) rs += __shfl_xor(rs, off, 32);
      lrow[v] = lrow[v] * alpha + rs;
      O[0][v] *= alpha; O[1][v] *= alpha; O[2][v] *= alpha; O[3][v] *= alpha;
      int prow = v + 8 * g;           // C-layout row
      f16* pr_ = Pw + prow * 72 + ln;
      pr_[0]  = (f16)p0;
      pr_[16] = (f16)p1;
      pr_[32] = (f16)p2;
      pr_[48] = (f16)p3;
    }
    asm volatile("s_wait_dscnt 0" ::: "memory");

    // O += P V : two key-halves of 32, four 16-dim output tiles
    v16h Ap0 = pack16(Pw + ln * 72 + g * 16,      Pw + ln * 72 + g * 16 + 8);
    v16h Ap1 = pack16(Pw + ln * 72 + 32 + g * 16, Pw + ln * 72 + 32 + g * 16 + 8);
#pragma unroll
    for (int dt = 0; dt < 4; dt++) {
      const f16* vr = VshT + (dt * 16 + ln) * 72;
      v16h Bv = pack16(vr + g * 16, vr + g * 16 + 8);
      O[dt] = __builtin_amdgcn_wmma_f32_16x16x32_f16(false, Ap0, false, Bv, (short)0, O[dt], false, false);
      Bv = pack16(vr + 32 + g * 16, vr + 32 + g * 16 + 8);
      O[dt] = __builtin_amdgcn_wmma_f32_16x16x32_f16(false, Ap1, false, Bv, (short)0, O[dt], false, false);
    }
    __syncthreads();
  }

  // normalize and store o[b][n][h][d]
#pragma unroll
  for (int v = 0; v < 8; v++) {
    float inv = 1.f / lrow[v];
    int row = q0 + v + 8 * g;
    size_t base = (((size_t)b * NTOK + row) * NHD + h) * 64;
#pragma unroll
    for (int dt = 0; dt < 4; dt++) o[base + dt * 16 + ln] = O[dt][v] * inv;
  }
}

// ---------------- attn output proj + residual + LN1 + LN2 -> spatial ----------------
__global__ __launch_bounds__(64) void k_attn_out(const float* __restrict__ o,
    const float* __restrict__ Wo, const float* __restrict__ bo,
    const float* __restrict__ x,
    const float* __restrict__ g1, const float* __restrict__ b1,
    const float* __restrict__ g2, const float* __restrict__ b2,
    float* __restrict__ spatial) {
  __shared__ float red[64];
  int pix = blockIdx.x, t = threadIdx.x;
  const float* orow = o + (size_t)pix * 512;
  float acc = bo[t];
#pragma unroll 4
  for (int k = 0; k < 512; k++) acc += orow[k] * Wo[(size_t)k * 64 + t];
  float s = x[(size_t)pix * 64 + t] + acc;
  float m  = reduce64(s, red, t) * (1.f / 64.f);
  float s2 = reduce64(s * s, red, t) * (1.f / 64.f);
  float x1 = (s - m) * rsqrtf(s2 - m * m + 1e-3f) * g1[t] + b1[t];
  float m2 = reduce64(x1, red, t) * (1.f / 64.f);
  float q2 = reduce64(x1 * x1, red, t) * (1.f / 64.f);
  spatial[(size_t)pix * 64 + t] = (x1 - m2) * rsqrtf(q2 - m2 * m2 + 1e-3f) * g2[t] + b2[t];
}

// ---------------- DCT / IDCT (48-pt matrix products) ----------------
__global__ void k_dct_rows(const float* __restrict__ x, const float* __restrict__ D, float* __restrict__ out) {
  int tid = blockIdx.x * blockDim.x + threadIdx.x;
  if (tid >= BB * HH * WW * CC) return;
  int c = tid & 63, w = (tid >> 6) % WW, k = ((tid >> 6) / WW) % HH, b = (tid >> 6) / (WW * HH);
  float acc = 0.f;
  for (int hh = 0; hh < HH; hh++)
    acc += D[k * 48 + hh] * x[(((size_t)b * HH + hh) * WW + w) * CC + c];
  out[tid] = acc;
}
__global__ void k_dct_cols(const float* __restrict__ t1, const float* __restrict__ D, float* __restrict__ out) {
  int tid = blockIdx.x * blockDim.x + threadIdx.x;
  if (tid >= BB * HH * WW * CC) return;
  int c = tid & 63, l = (tid >> 6) % WW, k = ((tid >> 6) / WW) % HH, b = (tid >> 6) / (WW * HH);
  float acc = 0.f;
  for (int w = 0; w < WW; w++)
    acc += D[l * 48 + w] * t1[(((size_t)b * HH + k) * WW + w) * CC + c];
  out[tid] = acc;
}
__global__ void k_idct_rows(const float* __restrict__ gi, const float* __restrict__ D, float* __restrict__ out) {
  int tid = blockIdx.x * blockDim.x + threadIdx.x;
  if (tid >= BB * HH * WW * CC) return;
  int c = tid & 63, w = (tid >> 6) % WW, hh = ((tid >> 6) / WW) % HH, b = (tid >> 6) / (WW * HH);
  float acc = 0.f;
  for (int k = 0; k < HH; k++)
    acc += D[k * 48 + hh] * gi[(((size_t)b * HH + k) * WW + w) * CC + c];
  out[tid] = acc;
}
__global__ void k_idct_cols(const float* __restrict__ t1, const float* __restrict__ D, float* __restrict__ out) {
  int tid = blockIdx.x * blockDim.x + threadIdx.x;
  if (tid >= BB * HH * WW * CC) return;
  int c = tid & 63, w = (tid >> 6) % WW, hh = ((tid >> 6) / WW) % HH, b = (tid >> 6) / (WW * HH);
  float acc = 0.f;
  for (int l = 0; l < WW; l++)
    acc += D[l * 48 + w] * t1[(((size_t)b * HH + hh) * WW + l) * CC + c];
  out[tid] = acc;
}

// ---------------- band features / weights / scaling ----------------
__global__ void k_band_feat(const float* __restrict__ d, float* __restrict__ bf) {
  int tid = blockIdx.x * blockDim.x + threadIdx.x;
  if (tid >= BB * NBANDS * CC) return;
  int c = tid & 63, k = (tid >> 6) & 7, b = tid >> 9;
  float acc = 0.f;
  for (int r = 0; r < 6; r++)
    for (int w = 0; w < WW; w++)
      acc += d[(((size_t)b * HH + k * 6 + r) * WW + w) * CC + c];
  bf[tid] = acc * (1.f / 288.f);
}
__global__ void k_band_bw(const float* __restrict__ bf, const float* __restrict__ W,
                          const float* __restrict__ bias, float* __restrict__ bw) {
  __shared__ float lg[8];
  int b = blockIdx.x, j = threadIdx.x;
  float acc = bias[j];
  for (int i = 0; i < 512; i++) acc += bf[b * 512 + i] * W[i * 8 + j];
  lg[j] = acc; __syncthreads();
  float mx = lg[0];
  for (int i = 1; i < 8; i++) mx = fmaxf(mx, lg[i]);
  float den = 0.f;
  for (int i = 0; i < 8; i++) den += __expf(lg[i] - mx);
  bw[b * 8 + j] = __expf(acc - mx) / den;
}
__global__ void k_scale_bands(float* __restrict__ d, const float* __restrict__ bw) {
  int tid = blockIdx.x * blockDim.x + threadIdx.x;
  if (tid >= BB * HH * WW * CC) return;
  int k = ((tid >> 6) / WW) % HH, b = (tid >> 6) / (WW * HH);
  d[tid] *= bw[b * 8 + k / 6];
}

// ---------------- phase reduction ----------------
__global__ void k_phase_partial(const float* __restrict__ x, const float* __restrict__ W,
                                float* __restrict__ part) {
  __shared__ float red[256];
  int b = blockIdx.x / 72, seg = blockIdx.x % 72;
  int base = seg * 2048 + threadIdx.x;
  float acc = 0.f;
#pragma unroll
  for (int i = 0; i < 8; i++) {
    int idx = base + i * 256;
    float xv = x[(size_t)b * 147456 + idx];
    acc += (xv < 0.f ? PI_F : 0.f) * W[idx];
  }
  red[threadIdx.x] = acc; __syncthreads();
  for (int off = 128; off >= 1; off >>= 1) {
    if (threadIdx.x < off) red[threadIdx.x] += red[threadIdx.x + off];
    __syncthreads();
  }
  if (threadIdx.x == 0) part[blockIdx.x] = red[0];
}
__global__ void k_phase_final(const float* __restrict__ part, float* __restrict__ praw) {
  int b = threadIdx.x;
  if (b >= BB) return;
  float s = 0.f;
  for (int i = 0; i < 72; i++) s += part[b * 72 + i];
  praw[b] = s;
}

// ---------------- freq 3x3 conv (* sigmoid(phase)) ----------------
__global__ void k_conv_freq(const float* __restrict__ wd, const float* __restrict__ W,
                            const float* __restrict__ bias, const float* __restrict__ praw,
                            const float* __restrict__ pbias, float* __restrict__ out) {
  int tid = blockIdx.x * blockDim.x + threadIdx.x;
  if (tid >= BB * HH * WW * CC) return;
  int co = tid & 63, xw = (tid >> 6) % WW, y = ((tid >> 6) / WW) % HH, b = (tid >> 6) / (WW * HH);
  float acc = bias[co];
  for (int ky = 0; ky < 3; ky++) {
    int yy = y + ky - 1; if ((unsigned)yy >= (unsigned)HH) continue;
    for (int kx = 0; kx < 3; kx++) {
      int xx = xw + kx - 1; if ((unsigned)xx >= (unsigned)WW) continue;
      const float* ip = wd + (((size_t)b * HH + yy) * WW + xx) * CC;
      const float* wp = W + (size_t)(ky * 3 + kx) * 64 * 64 + co;
      for (int ci = 0; ci < 64; ci++) acc += ip[ci] * wp[ci * 64];
    }
  }
  float pw = 1.f / (1.f + __expf(-(praw[b] + pbias[0])));
  out[tid] = acc * pw;
}

// ---------------- wavelet ----------------
__global__ void k_extract(const float* __restrict__ x, float* __restrict__ ll0) {
  int tid = blockIdx.x * blockDim.x + threadIdx.x;
  if (tid >= BB * HH * WW) return;
  ll0[tid] = x[(size_t)tid * 64];
}
__global__ void k_haar(const float* __restrict__ src, int n,
                       float* __restrict__ ll, float* __restrict__ lh,
                       float* __restrict__ hl, float* __restrict__ hh, int total) {
  int tid = blockIdx.x * blockDim.x + threadIdx.x;
  if (tid >= total) return;
  int hn = n >> 1;
  int j = tid % hn, i = (tid / hn) % hn, b = tid / (hn * hn);
  const float* s = src + (size_t)b * n * n;
  float a = s[(2 * i) * n + 2 * j];
  float bq = s[(2 * i) * n + 2 * j + 1];
  float c = s[(2 * i + 1) * n + 2 * j];
  float dq = s[(2 * i + 1) * n + 2 * j + 1];
  ll[tid] = (a + bq + c + dq) * 0.5f;
  lh[tid] = (a - bq + c - dq) * 0.5f;
  hl[tid] = (a + bq - c - dq) * 0.5f;
  hh[tid] = (a - bq - c + dq) * 0.5f;
}
__global__ void k_detw(const float* __restrict__ hb, const float* __restrict__ vb,
                       const float* __restrict__ db, int s,
                       const float* __restrict__ W, const float* __restrict__ bias,
                       float* __restrict__ logits) {
  int b = blockIdx.x;
  int wv = threadIdx.x >> 5, ln = threadIdx.x & 31;
  int m2 = s * s, len = 3 * m2;
  float acc = 0.f;
  for (int i = ln; i < len; i += 32) {
    int seg = i / m2, off = i % m2;
    const float* src = (seg == 0) ? hb : (seg == 1 ? vb : db);
    acc += src[b * m2 + off] * W[i * 3 + wv];
  }
#pragma unroll
  for (int off = 16; off >= 1; off >>= 1) acc += __shfl_xor(acc, off, 32);
  if (ln == 0) logits[b * 3 + wv] = acc + bias[wv];
}
__global__ __launch_bounds__(64) void k_rec0(const float* __restrict__ ll3,
                                             const float* __restrict__ cw,
                                             const float* __restrict__ cb,
                                             float* __restrict__ rec) {
  __shared__ float red[64];
  __shared__ float awsh;
  int b = blockIdx.x, t = threadIdx.x;
  float p = (t < 36) ? ll3[b * 36 + t] * cw[t] : 0.f;
  red[t] = p; __syncthreads();
  for (int off = 32; off >= 1; off >>= 1) {
    if (t < off) red[t] += red[t + off];
    __syncthreads();
  }
  if (t == 0) awsh = 1.f / (1.f + __expf(-(red[0] + cb[0])));
  __syncthreads();
  if (t < 36) rec[b * 36 + t] = ll3[b * 36 + t] * awsh;
}
__global__ void k_idwt(const float* __restrict__ rec, const float* __restrict__ hb,
                       const float* __restrict__ vb, const float* __restrict__ db,
                       const float* __restrict__ logits, int s,
                       float* __restrict__ out, int total) {
  int tid = blockIdx.x * blockDim.x + threadIdx.x;
  if (tid >= total) return;
  int j = tid % s, i = (tid / s) % s, b = tid / (s * s);
  float l0 = logits[b * 3], l1 = logits[b * 3 + 1], l2 = logits[b * 3 + 2];
  float mx = fmaxf(l0, fmaxf(l1, l2));
  float e0 = __expf(l0 - mx), e1 = __expf(l1 - mx), e2 = __expf(l2 - mx);
  float den = e0 + e1 + e2;
  float w0 = e0 / den, w1 = e1 / den, w2 = e2 / den;
  int m2 = s * s;
  float ll = rec[b * m2 + i * s + j];
  float lh = hb[b * m2 + i * s + j] * w0;
  float hl = vb[b * m2 + i * s + j] * w1;
  float hh = db[b * m2 + i * s + j] * w2;
  int n2 = 2 * s;
  float* ob = out + (size_t)b * n2 * n2;
  ob[(2 * i) * n2 + 2 * j]         = (ll + lh + hl + hh) * 0.5f;
  ob[(2 * i) * n2 + 2 * j + 1]     = (ll - lh + hl - hh) * 0.5f;
  ob[(2 * i + 1) * n2 + 2 * j]     = (ll + lh - hl - hh) * 0.5f;
  ob[(2 * i + 1) * n2 + 2 * j + 1] = (ll - lh - hl + hh) * 0.5f;
}

// ---------------- feat 3x3 conv over concat(spatial,freq,wav) ----------------
__global__ void k_feat_conv(const float* __restrict__ sp, const float* __restrict__ fq,
                            const float* __restrict__ wav, const float* __restrict__ W,
                            const float* __restrict__ bias, float* __restrict__ out) {
  int tid = blockIdx.x * blockDim.x + threadIdx.x;
  if (tid >= BB * HH * WW * 64) return;
  int co = tid & 63, xw = (tid >> 6) % WW, y = ((tid >> 6) / WW) % HH, b = (tid >> 6) / (WW * HH);
  float acc = bias[co];
  for (int ky = 0; ky < 3; ky++) {
    int yy = y + ky - 1; if ((unsigned)yy >= (unsigned)HH) continue;
    for (int kx = 0; kx < 3; kx++) {
      int xx = xw + kx - 1; if ((unsigned)xx >= (unsigned)WW) continue;
      size_t pix = ((size_t)b * HH + yy) * WW + xx;
      const float* wp = W + (size_t)(ky * 3 + kx) * 129 * 64 + co;
      const float* s0 = sp + pix * 64;
      const float* s1 = fq + pix * 64;
      for (int ci = 0; ci < 64; ci++) acc += s0[ci] * wp[ci * 64];
      for (int ci = 0; ci < 64; ci++) acc += s1[ci] * wp[(64 + ci) * 64];
      acc += wav[pix] * wp[128 * 64];
    }
  }
  out[tid] = acc;
}

// ---------------- LN(feat) + 1x1 conv to 32 ----------------
__global__ __launch_bounds__(64) void k_final(const float* __restrict__ yp,
    const float* __restrict__ g, const float* __restrict__ bb,
    const float* __restrict__ Wout, const float* __restrict__ bout,
    float* __restrict__ out) {
  __shared__ float red[64];
  __shared__ float ysh[64];
  int pix = blockIdx.x, t = threadIdx.x;
  float v = yp[(size_t)pix * 64 + t];
  float m  = reduce64(v, red, t) * (1.f / 64.f);
  float s2 = reduce64(v * v, red, t) * (1.f / 64.f);
  float y = (v - m) * rsqrtf(s2 - m * m + 1e-3f) * g[t] + bb[t];
  ysh[t] = y; __syncthreads();
  if (t < 32) {
    float acc = bout[t];
    for (int c = 0; c < 64; c++) acc += ysh[c] * Wout[c * 32 + t];
    out[(size_t)pix * 32 + t] = acc;
  }
}

// ---------------- host launcher ----------------
extern "C" void kernel_launch(void* const* d_in, const int* in_sizes, int n_in,
                              void* d_out, int out_size, void* d_ws, size_t ws_size,
                              hipStream_t stream) {
  (void)in_sizes; (void)n_in; (void)out_size; (void)ws_size;
  const float* x      = (const float*)d_in[0];
  const float* pos_W  = (const float*)d_in[1];
  const float* pos_b  = (const float*)d_in[2];
  const float* Wq     = (const float*)d_in[3];
  const float* bq     = (const float*)d_in[4];
  const float* Wk     = (const float*)d_in[5];
  const float* bk     = (const float*)d_in[6];
  const float* Wv     = (const float*)d_in[7];
  const float* bv     = (const float*)d_in[8];
  const float* Wo     = (const float*)d_in[9];
  const float* bo     = (const float*)d_in[10];
  const float* ln1_g  = (const float*)d_in[11];
  const float* ln1_b  = (const float*)d_in[12];
  const float* ln2_g  = (const float*)d_in[13];
  const float* ln2_b  = (const float*)d_in[14];
  const float* band_W = (const float*)d_in[15];
  const float* band_b = (const float*)d_in[16];
  const float* freq_W = (const float*)d_in[17];
  const float* freq_b = (const float*)d_in[18];
  const float* phase_W= (const float*)d_in[19];
  const float* phase_b= (const float*)d_in[20];
  const float* coeff_W= (const float*)d_in[21];
  const float* coeff_b= (const float*)d_in[22];
  const float* dW1    = (const float*)d_in[23];
  const float* db1    = (const float*)d_in[24];
  const float* dW2    = (const float*)d_in[25];
  const float* db2    = (const float*)d_in[26];
  const float* dW3    = (const float*)d_in[27];
  const float* db3    = (const float*)d_in[28];
  const float* feat_W = (const float*)d_in[29];
  const float* feat_b = (const float*)d_in[30];
  const float* lnf_g  = (const float*)d_in[31];
  const float* lnf_b  = (const float*)d_in[32];
  const float* out_W  = (const float*)d_in[33];
  const float* out_b  = (const float*)d_in[34];

  char* ws = (char*)d_ws;
  float* Dm   = (float*)(ws + o_dct);
  float* qbuf = (float*)(ws + o_q);
  f16*   qh   = (f16*)(ws + o_qh);
  f16*   kh   = (f16*)(ws + o_kh);
  f16*   vhT  = (f16*)(ws + o_vh);
  float* obuf = (float*)(ws + o_o);
  float* t1   = (float*)(ws + o_t1);
  float* t2   = (float*)(ws + o_t2);
  float* t3   = (float*)(ws + o_t3);
  float* bf   = (float*)(ws + o_bf);
  float* bw   = (float*)(ws + o_bw);
  float* pp   = (float*)(ws + o_pp);
  float* pr   = (float*)(ws + o_pr);
  float* ll0  = (float*)(ws + o_ll0);
  float* ll1  = (float*)(ws + o_ll1);
  float* h1   = (float*)(ws + o_h1);
  float* v1   = (float*)(ws + o_v1);
  float* d1   = (float*)(ws + o_d1);
  float* ll2  = (float*)(ws + o_ll2);
  float* h2   = (float*)(ws + o_h2);
  float* v2   = (float*)(ws + o_v2);
  float* d2   = (float*)(ws + o_d2);
  float* ll3  = (float*)(ws + o_ll3);
  float* h3   = (float*)(ws + o_h3);
  float* v3   = (float*)(ws + o_v3);
  float* d3   = (float*)(ws + o_d3);
  float* r6   = (float*)(ws + o_r6);
  float* r12  = (float*)(ws + o_r12);
  float* r24  = (float*)(ws + o_r24);
  float* lg1  = (float*)(ws + o_lg1);
  float* lg2  = (float*)(ws + o_lg2);
  float* lg3  = (float*)(ws + o_lg3);
  float* yp   = (float*)(ws + o_yp);

  float* out0 = (float*)d_out;               // (B,H,W,32)
  float* spat = out0 + 147456;               // (B,H,W,64)
  float* frq  = out0 + 442368;               // (B,H,W,64)
  float* wav  = out0 + 737280;               // (B,H,W,1)

  // DCT basis
  k_dctmat<<<9, 256, 0, stream>>>(Dm);
  // attention path
  k_pos<<<1152, 256, 0, stream>>>(x, pos_W, pos_b, qbuf);
  k_proj<<<9216, 256, 0, stream>>>(qbuf, Wq, bq, qh, 0.125f);   // fold 1/sqrt(KD)
  k_proj<<<9216, 256, 0, stream>>>(x,    Wk, bk, kh, 1.0f);
  k_projT<<<9216, 256, 0, stream>>>(x,   Wv, bv, vhT);          // [b][h][d][n]
  k_flash_attn<<<576, 128, 0, stream>>>(qh, kh, vhT, obuf);
  k_attn_out<<<4608, 64, 0, stream>>>(obuf, Wo, bo, x, ln1_g, ln1_b, ln2_g, ln2_b, spat);
  // DCT / band / phase / freq path
  k_dct_rows<<<1152, 256, 0, stream>>>(x, Dm, t1);
  k_dct_cols<<<1152, 256, 0, stream>>>(t1, Dm, t2);
  k_band_feat<<<4, 256, 0, stream>>>(t2, bf);
  k_band_bw<<<2, 8, 0, stream>>>(bf, band_W, band_b, bw);
  k_phase_partial<<<144, 256, 0, stream>>>(x, phase_W, pp);
  k_phase_final<<<1, 2, 0, stream>>>(pp, pr);
  k_scale_bands<<<1152, 256, 0, stream>>>(t2, bw);
  k_conv_freq<<<1152, 256, 0, stream>>>(t2, freq_W, freq_b, pr, phase_b, t3);
  k_idct_rows<<<1152, 256, 0, stream>>>(t3, Dm, t1);
  k_idct_cols<<<1152, 256, 0, stream>>>(t1, Dm, frq);
  // wavelet path
  k_extract<<<18, 256, 0, stream>>>(x, ll0);
  k_haar<<<5, 256, 0, stream>>>(ll0, 48, ll1, h1, v1, d1, BB * 24 * 24);
  k_haar<<<2, 256, 0, stream>>>(ll1, 24, ll2, h2, v2, d2, BB * 12 * 12);
  k_haar<<<1, 128, 0, stream>>>(ll2, 12, ll3, h3, v3, d3, BB * 6 * 6);
  k_detw<<<2, 96, 0, stream>>>(h3, v3, d3, 6,  dW1, db1, lg1);
  k_detw<<<2, 96, 0, stream>>>(h2, v2, d2, 12, dW2, db2, lg2);
  k_detw<<<2, 96, 0, stream>>>(h1, v1, d1, 24, dW3, db3, lg3);
  k_rec0<<<2, 64, 0, stream>>>(ll3, coeff_W, coeff_b, r6);
  k_idwt<<<1, 128, 0, stream>>>(r6,  h3, v3, d3, lg1, 6,  r12, BB * 6 * 6);
  k_idwt<<<2, 256, 0, stream>>>(r12, h2, v2, d2, lg2, 12, r24, BB * 12 * 12);
  k_idwt<<<5, 256, 0, stream>>>(r24, h1, v1, d1, lg3, 24, wav, BB * 24 * 24);
  // fuse + output
  k_feat_conv<<<1152, 256, 0, stream>>>(spat, frq, wav, feat_W, feat_b, yp);
  k_final<<<4608, 64, 0, stream>>>(yp, lnf_g, lnf_b, out_W, out_b, out0);
}
